// Net3D_56659208569398
// MI455X (gfx1250) — compile-verified, matrix operand order checked by reading, not run on hardware
//
#include <hip/hip_runtime.h>

// ---------------- problem constants ----------------
static constexpr int NN = 20000;   // nodes
static constexpr int NE = 320000;  // edges
static constexpr int HD = 128;     // hidden
static constexpr int NG = 64;      // graphs
static constexpr int NL = 4;       // mp layers
static constexpr int TG = 32;      // target

// ---------------- WMMA types / helpers ----------------
typedef __attribute__((ext_vector_type(16))) __bf16 v16bf;
typedef __attribute__((ext_vector_type(8)))  float  v8f;

union Frag {
  v16bf v;
  unsigned short u[16];
  uint4 q[2];
};

__device__ __forceinline__ unsigned short f2bf(float f) {
  unsigned u = __float_as_uint(f);
  unsigned r = u + 0x7FFFu + ((u >> 16) & 1u);   // round-to-nearest-even
  return (unsigned short)(r >> 16);
}
__device__ __forceinline__ float bf2f(unsigned short s) {
  return __uint_as_float(((unsigned)s) << 16);
}
// per-lane K index for 16-bit A/B fragments (ISA 7.12.2):
// halves 0..7  -> K = koff + 0..7
// halves 8..15 -> K = 16 + koff + 0..7        (koff = (lane&16)>>1)
__device__ __forceinline__ int kmap(int lane, int h) {
  return h + (h & 8) + ((lane & 16) >> 1);
}
// A fragment from a contiguous bf16 row: exactly two 16-byte loads.
__device__ __forceinline__ Frag loadA(const unsigned short* __restrict__ row, int koff) {
  Frag a;
  a.q[0] = *reinterpret_cast<const uint4*>(row + koff);
  a.q[1] = *reinterpret_cast<const uint4*>(row + 16 + koff);
  return a;
}
// A fragment from LDS bf16 row (lowers to 2x ds_load_b128).
__device__ __forceinline__ Frag loadA_lds(const unsigned short* row, int koff) {
  Frag a;
  a.q[0] = *reinterpret_cast<const uint4*>(row + koff);
  a.q[1] = *reinterpret_cast<const uint4*>(row + 16 + koff);
  return a;
}
__device__ __forceinline__ v8f wmma_bf16(const Frag& a, const Frag& b, v8f c) {
  return __builtin_amdgcn_wmma_f32_16x16x32_bf16(
      /*neg_a=*/false, a.v, /*neg_b=*/false, b.v,
      /*c_mod=*/(short)0, c, /*reuse_a=*/false, /*reuse_b=*/false);
}
__device__ __forceinline__ Frag loadB(const unsigned short* __restrict__ Ws,
                                      int frag, int lane) {
  Frag b;
  const uint4* p = reinterpret_cast<const uint4*>(Ws + ((size_t)frag * 32 + lane) * 16);
  b.q[0] = p[0];
  b.q[1] = p[1];
  return b;
}
__device__ __forceinline__ float siluf(float x) { return x / (1.f + __expf(-x)); }

// ---------------- weight pre-swizzle: f32 [K,N] -> bf16 WMMA B fragments ----------------
__global__ void swizzle_weights(const float* __restrict__ W,
                                unsigned short* __restrict__ out, int K, int Nc) {
  int idx = blockIdx.x * blockDim.x + threadIdx.x;
  if (idx >= K * Nc) return;
  int h    = idx & 15;
  int lane = (idx >> 4) & 31;
  int f    = idx >> 9;          // 512 halves per 32x16 fragment
  int NT   = Nc >> 4;
  int kt = f / NT, nt = f - kt * NT;
  int k = kt * 32 + kmap(lane, h);
  int n = nt * 16 + (lane & 15);
  out[idx] = f2bf(W[(size_t)k * Nc + n]);
}

// ---------------- init h (f32) and its bf16 mirror ----------------
__global__ void h_init_kernel(const float* __restrict__ emb, float* __restrict__ h,
                              unsigned short* __restrict__ hb, int total) {
  int i = blockIdx.x * blockDim.x + threadIdx.x;
  if (i < total) {
    const float v = emb[i & (HD - 1)];
    h[i] = v;
    hb[i] = f2bf(v);
  }
}

// ---------------- edge encoder: fourier(9) @ [9,128] + double silu -> bf16 ----------------
__global__ void edge_encode_kernel(const float* __restrict__ dist,
                                   const float* __restrict__ eW, const float* __restrict__ eb,
                                   unsigned short* __restrict__ e, int E) {
  int i = blockIdx.x * blockDim.x + threadIdx.x;
  if (i >= E) return;
  float d = dist[i];
  float f[9];
#pragma unroll
  for (int j = 0; j < 4; ++j) {
    float s = d / (float)(1 << j);
    f[j]     = __sinf(s);
    f[4 + j] = __cosf(s);
  }
  f[8] = d;
  unsigned short* ep = e + (size_t)i * HD;
  for (int c = 0; c < HD; ++c) {
    float a = eb[c];
#pragma unroll
    for (int j = 0; j < 9; ++j) a += f[j] * eW[j * HD + c];
    a = siluf(siluf(a));
    ep[c] = f2bf(a);
  }
}

// ---------------- edge message MLP + soft gate + scatter, one wave per 16 edges ----------------
__global__ void __launch_bounds__(128) edge_mlp_kernel(
    const unsigned short* __restrict__ hb,   // bf16 mirror of h [NN,128]
    unsigned short* __restrict__ e,          // bf16 edge features [NE,128], updated in place
    float* __restrict__ agg,
    const int* __restrict__ srcI, const int* __restrict__ dstI,
    const unsigned short* __restrict__ W1s, const float* __restrict__ b1,
    const unsigned short* __restrict__ W2s, const float* __restrict__ b2,
    const float* __restrict__ sW, const float* __restrict__ sb) {
  __shared__ unsigned short sT1[4][16 * HD];  // silu(m_in@W1) in bf16
  __shared__ float          sM [4][16 * HD];  // final message m (f32)
  const int wave = threadIdx.x >> 5, lane = threadIdx.x & 31;
  const int lo = lane & 15, hi = lane >> 4;
  const int koff = hi << 3;                   // 0 or 8
  const int tile = blockIdx.x * 4 + wave;
  const int e0 = tile * 16;
  const int edge = e0 + lo;

  // warm W2 into caches while phase-1 runs (global_prefetch_b8)
  __builtin_prefetch(W2s + (size_t)threadIdx.x * 128, 0, 3);

  const v8f z = {0.f, 0.f, 0.f, 0.f, 0.f, 0.f, 0.f, 0.f};
  v8f acc[8];
#pragma unroll
  for (int nt = 0; nt < 8; ++nt) acc[nt] = z;

  const int sN = srcI[edge], dN = dstI[edge];
  const unsigned short* hsrc = hb + (size_t)sN * HD;
  const unsigned short* hdst = hb + (size_t)dN * HD;
  const unsigned short* erow = e + (size_t)edge * HD;

  // ---- m_in[16,384] @ W1[384,128]: 12 K-chunks x 8 N-tiles ----
  // B fragments staged in an 8-wide register array -> one wide load clause per
  // K-chunk with incremental s_wait_loadcnt; A fragment software-pipelined.
  Frag a = loadA(hsrc, koff);
#pragma unroll
  for (int kt = 0; kt < 12; ++kt) {
    Frag bfr[8];
#pragma unroll
    for (int nt = 0; nt < 8; ++nt) bfr[nt] = loadB(W1s, kt * 8 + nt, lane);
    Frag an;
    if (kt < 11) {
      const int kn = kt + 1;
      const unsigned short* arow =
          (kn < 4) ? (hsrc + kn * 32) : (kn < 8) ? (hdst + (kn - 4) * 32) : (erow + (kn - 8) * 32);
      an = loadA(arow, koff);
    }
#pragma unroll
    for (int nt = 0; nt < 8; ++nt) acc[nt] = wmma_bf16(a, bfr[nt], acc[nt]);
    if (kt < 11) a = an;
  }

  unsigned short* t1 = sT1[wave];
#pragma unroll
  for (int nt = 0; nt < 8; ++nt) {
    const float bia = b1[nt * 16 + lo];
#pragma unroll
    for (int r = 0; r < 8; ++r)
      t1[(r + 8 * hi) * HD + nt * 16 + lo] = f2bf(siluf(acc[nt][r] + bia));
  }

  // ---- t1[16,128] @ W2[128,128]: 4 K-chunks x 8 N-tiles ----
  v8f acc2[8];
#pragma unroll
  for (int nt = 0; nt < 8; ++nt) acc2[nt] = z;
  Frag a2 = loadA_lds(t1 + lo * HD, koff);
#pragma unroll
  for (int kt = 0; kt < 4; ++kt) {
    Frag bfr[8];
#pragma unroll
    for (int nt = 0; nt < 8; ++nt) bfr[nt] = loadB(W2s, kt * 8 + nt, lane);
    Frag an;
    if (kt < 3) an = loadA_lds(t1 + lo * HD + (kt + 1) * 32, koff);
#pragma unroll
    for (int nt = 0; nt < 8; ++nt) acc2[nt] = wmma_bf16(a2, bfr[nt], acc2[nt]);
    if (kt < 3) a2 = an;
  }

  float* m = sM[wave];
#pragma unroll
  for (int nt = 0; nt < 8; ++nt) {
    const float bia = b2[nt * 16 + lo];
#pragma unroll
    for (int r = 0; r < 8; ++r)
      m[(r + 8 * hi) * HD + nt * 16 + lo] = siluf(acc2[nt][r] + bia);
  }

  // ---- epilogue: soft gate, e += m, agg[dst] += m*w ----
  const float sbv = sb[0];
  for (int row = 0; row < 16; ++row) {
    const float* mr = m + row * HD;
    float p = 0.f;
#pragma unroll
    for (int c = lane; c < HD; c += 32) p += mr[c] * sW[c];
#pragma unroll
    for (int off = 16; off > 0; off >>= 1) p += __shfl_xor(p, off, 32);
    const float w = 1.f / (1.f + __expf(-(p + sbv)));
    const int er = e0 + row;
    const int dn = dstI[er];
    unsigned short* ep = e + (size_t)er * HD;
    float* ap = agg + (size_t)dn * HD;
#pragma unroll
    for (int c = lane; c < HD; c += 32) {
      const float mv = mr[c];
      atomicAdd(ap + c, mv * w);                // L2-resident scatter
      ep[c] = f2bf(bf2f(ep[c]) + mv);           // residual edge update in bf16
    }
  }
}

// ---------------- node MLP (mode 0: h += upd MLP(agg+h); mode 1: hout = out MLP(h)) ----------------
__global__ void __launch_bounds__(128) node_mlp_kernel(
    float* __restrict__ h, unsigned short* __restrict__ hb,
    const float* __restrict__ agg,
    const unsigned short* __restrict__ W1s, const float* __restrict__ b1,
    const unsigned short* __restrict__ W2s, const float* __restrict__ b2,
    const int mode, float* __restrict__ hout, const int ntiles) {
  __shared__ unsigned short sT1[4][16 * HD];
  const int wave = threadIdx.x >> 5, lane = threadIdx.x & 31;
  const int lo = lane & 15, hi = lane >> 4;
  const int koff = hi << 3;
  const int tile = blockIdx.x * 4 + wave;
  if (tile >= ntiles) return;  // wave-uniform exit keeps EXEC all-ones for WMMA
  const int n0 = tile * 16;
  const int node = n0 + lo;

  __builtin_prefetch(W2s + (size_t)threadIdx.x * 128, 0, 3);

  const v8f z = {0.f, 0.f, 0.f, 0.f, 0.f, 0.f, 0.f, 0.f};
  v8f acc[8];
#pragma unroll
  for (int nt = 0; nt < 8; ++nt) acc[nt] = z;

#pragma unroll
  for (int kt = 0; kt < 4; ++kt) {
    Frag a;
    if (mode == 0) {  // input = h + agg (f32 residual path), convert on the fly
      const size_t base = (size_t)node * HD + kt * 32;
#pragma unroll
      for (int t = 0; t < 16; ++t) {
        const int k = kmap(lane, t);
        a.u[t] = f2bf(h[base + k] + agg[base + k]);
      }
    } else {          // input = h, bf16 mirror, 2x b128 loads
      a = loadA(hb + (size_t)node * HD + kt * 32, koff);
    }
    Frag bfr[8];
#pragma unroll
    for (int nt = 0; nt < 8; ++nt) bfr[nt] = loadB(W1s, kt * 8 + nt, lane);
#pragma unroll
    for (int nt = 0; nt < 8; ++nt) acc[nt] = wmma_bf16(a, bfr[nt], acc[nt]);
  }

  unsigned short* t1 = sT1[wave];
#pragma unroll
  for (int nt = 0; nt < 8; ++nt) {
    const float bia = b1[nt * 16 + lo];
#pragma unroll
    for (int r = 0; r < 8; ++r)
      t1[(r + 8 * hi) * HD + nt * 16 + lo] = f2bf(siluf(acc[nt][r] + bia));
  }

  v8f acc2[8];
#pragma unroll
  for (int nt = 0; nt < 8; ++nt) acc2[nt] = z;
  Frag a2 = loadA_lds(t1 + lo * HD, koff);
#pragma unroll
  for (int kt = 0; kt < 4; ++kt) {
    Frag bfr[8];
#pragma unroll
    for (int nt = 0; nt < 8; ++nt) bfr[nt] = loadB(W2s, kt * 8 + nt, lane);
    Frag an;
    if (kt < 3) an = loadA_lds(t1 + lo * HD + (kt + 1) * 32, koff);
#pragma unroll
    for (int nt = 0; nt < 8; ++nt) acc2[nt] = wmma_bf16(a2, bfr[nt], acc2[nt]);
    if (kt < 3) a2 = an;
  }

#pragma unroll
  for (int nt = 0; nt < 8; ++nt) {
    const float bia = b2[nt * 16 + lo];
#pragma unroll
    for (int r = 0; r < 8; ++r) {
      const size_t idx = (size_t)(n0 + r + 8 * hi) * HD + nt * 16 + lo;
      const float x = acc2[nt][r] + bia;      // last_activation = None
      if (mode == 0) {
        const float hv = h[idx] + x;          // residual node update
        h[idx] = hv;
        hb[idx] = f2bf(hv);                   // refresh bf16 mirror
      } else {
        hout[idx] = x;
      }
    }
  }
}

// ---------------- graph readout (batch is the known sorted formula) ----------------
__global__ void readout_kernel(const float* __restrict__ hout, float* __restrict__ ro) {
  const int g = blockIdx.x, c = threadIdx.x;
  const int start = (g * NN + NG - 1) / NG;
  const int end   = ((g + 1) * NN + NG - 1) / NG;
  float s = 0.f, mx = -3.402823466e38f;
  for (int i = start; i < end; ++i) {
    const float v = hout[(size_t)i * HD + c];
    s += v;
    mx = fmaxf(mx, v);
  }
  ro[(size_t)g * 384 + c]       = s / (float)(end - start);  // mean
  ro[(size_t)g * 384 + 128 + c] = s;                          // sum
  ro[(size_t)g * 384 + 256 + c] = mx;                         // max
}

// ---------------- readout head: [64,384]@[384,128] relu @[128,32], one block ----------------
__global__ void __launch_bounds__(128) readout_mlp_kernel(
    const float* __restrict__ ro,
    const unsigned short* __restrict__ W1s, const float* __restrict__ b1,
    const unsigned short* __restrict__ W2s, const float* __restrict__ b2,
    float* __restrict__ out) {
  __shared__ unsigned short sT1[4][16 * HD];
  const int wave = threadIdx.x >> 5, lane = threadIdx.x & 31;
  const int lo = lane & 15, hi = lane >> 4;
  const int koff = hi << 3;
  const int g0 = wave * 16;
  const int row = g0 + lo;

  const v8f z = {0.f, 0.f, 0.f, 0.f, 0.f, 0.f, 0.f, 0.f};
  v8f acc[8];
#pragma unroll
  for (int nt = 0; nt < 8; ++nt) acc[nt] = z;

#pragma unroll
  for (int kt = 0; kt < 12; ++kt) {
    Frag a;
    const float* rp = ro + (size_t)row * 384 + kt * 32;
#pragma unroll
    for (int t = 0; t < 16; ++t) a.u[t] = f2bf(rp[kmap(lane, t)]);
    Frag bfr[8];
#pragma unroll
    for (int nt = 0; nt < 8; ++nt) bfr[nt] = loadB(W1s, kt * 8 + nt, lane);
#pragma unroll
    for (int nt = 0; nt < 8; ++nt) acc[nt] = wmma_bf16(a, bfr[nt], acc[nt]);
  }

  unsigned short* t1 = sT1[wave];
#pragma unroll
  for (int nt = 0; nt < 8; ++nt) {
    const float bia = b1[nt * 16 + lo];
#pragma unroll
    for (int r = 0; r < 8; ++r)
      t1[(r + 8 * hi) * HD + nt * 16 + lo] = f2bf(fmaxf(acc[nt][r] + bia, 0.f));  // relu
  }

  v8f acc2[2];
  acc2[0] = z; acc2[1] = z;
#pragma unroll
  for (int kt = 0; kt < 4; ++kt) {
    const Frag a = loadA_lds(t1 + lo * HD + kt * 32, koff);
    Frag bfr[2];
#pragma unroll
    for (int nt = 0; nt < 2; ++nt) bfr[nt] = loadB(W2s, kt * 2 + nt, lane);
#pragma unroll
    for (int nt = 0; nt < 2; ++nt) acc2[nt] = wmma_bf16(a, bfr[nt], acc2[nt]);
  }

#pragma unroll
  for (int nt = 0; nt < 2; ++nt) {
    const float bia = b2[nt * 16 + lo];
#pragma unroll
    for (int r = 0; r < 8; ++r)
      out[(size_t)(g0 + r + 8 * hi) * TG + nt * 16 + lo] = acc2[nt][r] + bia;
  }
}

// ---------------- host-side orchestration ----------------
extern "C" void kernel_launch(void* const* d_in, const int* in_sizes, int n_in,
                              void* d_out, int out_size, void* d_ws, size_t ws_size,
                              hipStream_t stream) {
  const int*   edge_index = (const int*)d_in[1];
  const float* edge_attr  = (const float*)d_in[2];
  const float* node_emb   = (const float*)d_in[4];
  const float* edge_W     = (const float*)d_in[5];
  const float* edge_b     = (const float*)d_in[6];
  const float* msg_W1 = (const float*)d_in[7];
  const float* msg_b1 = (const float*)d_in[8];
  const float* msg_W2 = (const float*)d_in[9];
  const float* msg_b2 = (const float*)d_in[10];
  const float* soft_W = (const float*)d_in[11];
  const float* soft_b = (const float*)d_in[12];
  const float* upd_W1 = (const float*)d_in[13];
  const float* upd_b1 = (const float*)d_in[14];
  const float* upd_W2 = (const float*)d_in[15];
  const float* upd_b2 = (const float*)d_in[16];
  const float* no_W1  = (const float*)d_in[17];
  const float* no_b1  = (const float*)d_in[18];
  const float* no_W2  = (const float*)d_in[19];
  const float* no_b2  = (const float*)d_in[20];
  const float* out_W1 = (const float*)d_in[21];
  const float* out_b1 = (const float*)d_in[22];
  const float* out_W2 = (const float*)d_in[23];
  const float* out_b2 = (const float*)d_in[24];
  float* out = (float*)d_out;

  // ---- workspace carve (~119 MB total) ----
  char* p = (char*)d_ws;
  auto carve = [&](size_t bytes) {
    char* r = p;
    p += (bytes + 255) & ~(size_t)255;
    return r;
  };
  float* h    = (float*)carve((size_t)NN * HD * 4);
  float* agg  = (float*)carve((size_t)NN * HD * 4);
  float* hout = (float*)carve((size_t)NN * HD * 4);
  float* ro   = (float*)carve((size_t)NG * 384 * 4);
  unsigned short* hb      = (unsigned short*)carve((size_t)NN * HD * 2);
  unsigned short* e       = (unsigned short*)carve((size_t)NE * HD * 2);
  unsigned short* msg_W1s = (unsigned short*)carve((size_t)NL * 384 * HD * 2);
  unsigned short* msg_W2s = (unsigned short*)carve((size_t)NL * HD * HD * 2);
  unsigned short* upd_W1s = (unsigned short*)carve((size_t)NL * HD * HD * 2);
  unsigned short* upd_W2s = (unsigned short*)carve((size_t)NL * HD * HD * 2);
  unsigned short* no_W1s  = (unsigned short*)carve((size_t)HD * HD * 2);
  unsigned short* no_W2s  = (unsigned short*)carve((size_t)HD * HD * 2);
  unsigned short* out_W1s = (unsigned short*)carve((size_t)384 * HD * 2);
  unsigned short* out_W2s = (unsigned short*)carve((size_t)HD * TG * 2);

  auto swz = [&](const float* W, unsigned short* dst, int K, int Nc) {
    const int total = K * Nc;
    swizzle_weights<<<(total + 255) / 256, 256, 0, stream>>>(W, dst, K, Nc);
  };
  for (int l = 0; l < NL; ++l) {
    swz(msg_W1 + (size_t)l * 384 * HD, msg_W1s + (size_t)l * 384 * HD, 384, HD);
    swz(msg_W2 + (size_t)l * HD * HD,  msg_W2s + (size_t)l * HD * HD,  HD,  HD);
    swz(upd_W1 + (size_t)l * HD * HD,  upd_W1s + (size_t)l * HD * HD,  HD,  HD);
    swz(upd_W2 + (size_t)l * HD * HD,  upd_W2s + (size_t)l * HD * HD,  HD,  HD);
  }
  swz(no_W1, no_W1s, HD, HD);
  swz(no_W2, no_W2s, HD, HD);
  swz(out_W1, out_W1s, 384, HD);
  swz(out_W2, out_W2s, HD, TG);

  h_init_kernel<<<(NN * HD + 255) / 256, 256, 0, stream>>>(node_emb, h, hb, NN * HD);
  edge_encode_kernel<<<(NE + 255) / 256, 256, 0, stream>>>(edge_attr, edge_W, edge_b, e, NE);

  const int* srcI = edge_index;       // row 0: x_j (source)
  const int* dstI = edge_index + NE;  // row 1: x_i (target)
  const int node_tiles = NN / 16;     // 1250
  const int node_blocks = (node_tiles + 3) / 4;

  for (int l = 0; l < NL; ++l) {
    hipMemsetAsync(agg, 0, (size_t)NN * HD * 4, stream);
    edge_mlp_kernel<<<NE / 64, 128, 0, stream>>>(
        hb, e, agg, srcI, dstI,
        msg_W1s + (size_t)l * 384 * HD, msg_b1 + (size_t)l * HD,
        msg_W2s + (size_t)l * HD * HD,  msg_b2 + (size_t)l * HD,
        soft_W + (size_t)l * HD, soft_b + l);
    node_mlp_kernel<<<node_blocks, 128, 0, stream>>>(
        h, hb, agg,
        upd_W1s + (size_t)l * HD * HD, upd_b1 + (size_t)l * HD,
        upd_W2s + (size_t)l * HD * HD, upd_b2 + (size_t)l * HD,
        /*mode=*/0, nullptr, node_tiles);
  }

  node_mlp_kernel<<<node_blocks, 128, 0, stream>>>(
      h, hb, agg, no_W1s, no_b1, no_W2s, no_b2, /*mode=*/1, hout, node_tiles);

  readout_kernel<<<NG, HD, 0, stream>>>(hout, ro);
  readout_mlp_kernel<<<1, 128, 0, stream>>>(ro, out_W1s, out_b1, out_W2s, out_b2, out);
}
